// SelfCorAttention_68350109549088
// MI455X (gfx1250) — matrix operand order repaired and checked by reading, not compiled, not run on hardware
//
#include <hip/hip_runtime.h>

typedef __bf16 bf16;
typedef __attribute__((ext_vector_type(16))) __bf16 v16bf;
typedef __attribute__((ext_vector_type(8)))  float  v8f;

#define WMMA_BF16(a,b,c) \
  __builtin_amdgcn_wmma_f32_16x16x32_bf16(false,(a),false,(b),(short)0,(c),false,false)

// CDNA5 async direct global->LDS copy (16B per lane), tracked by ASYNCcnt.
__device__ __forceinline__ void async_b128(void* lds, const void* gbase, unsigned goff) {
  asm volatile("global_load_async_to_lds_b128 %0, %1, %2"
               :: "v"((unsigned)(uintptr_t)lds), "v"(goff), "s"(gbase)
               : "memory");
}
#define WAIT_ASYNC(imm) asm volatile("s_wait_asynccnt " imm ::: "memory")

__device__ __forceinline__ unsigned ldpack(const bf16* p) {
  union { bf16 h[2]; unsigned u; } t;
  t.h[0] = p[0]; t.h[1] = p[1];
  return t.u;
}
__device__ __forceinline__ unsigned pack2(bf16 lo, bf16 hi) {
  union { bf16 h[2]; unsigned u; } t;
  t.h[0] = lo; t.h[1] = hi;
  return t.u;
}

// A fragment (16x32 bf16) from row-major source: elem(m,k) at base[m*ld + k0 + k]
// Contiguous-k layout used here: lane m = L%16; lanes0-15 kb=0, lanes16-31 kb=16; VGPR j: k=kb+2j
__device__ __forceinline__ v16bf a_frag(const bf16* base, int ld, int k0, int lane) {
  union { v16bf v; unsigned u[8]; } f;
  int m  = lane & 15;
  int kb = (lane >> 4) * 16;
  const bf16* row = base + m * ld + k0;
#pragma unroll
  for (int j = 0; j < 4; ++j) f.u[j]     = ldpack(row + kb + 2 * j);
#pragma unroll
  for (int j = 0; j < 4; ++j) f.u[4 + j] = ldpack(row + 8 + kb + 2 * j);
  return f.v;
}

// B fragment (32x16 bf16) from source stored [n][k] row-major: elem(k,n) at base[n*ld + k0 + k]
// Contiguous-k layout used here: lane n = L%16; VGPR j: k = (L/16)*16 + 2j
__device__ __forceinline__ v16bf b_frag_nk(const bf16* base, int ld, int k0, int lane) {
  union { v16bf v; unsigned u[8]; } f;
  int n  = lane & 15;
  int kb = (lane >> 4) * 16;
  const bf16* row = base + n * ld + k0 + kb;
#pragma unroll
  for (int j = 0; j < 8; ++j) f.u[j] = ldpack(row + 2 * j);
  return f.v;
}

// B fragment (32x16 bf16) from source stored [k][n] row-major: elem(k,n) at base[(k0+k)*ld + n0 + n]
__device__ __forceinline__ v16bf b_frag_kn(const bf16* base, int ld, int k0, int n0, int lane) {
  union { v16bf v; unsigned u[8]; } f;
  int n  = n0 + (lane & 15);
  int kb = k0 + (lane >> 4) * 16;
#pragma unroll
  for (int j = 0; j < 8; ++j)
    f.u[j] = pack2(base[(kb + 2 * j) * ld + n], base[(kb + 2 * j + 1) * ld + n]);
  return f.v;
}

__global__ void cvt_bf16_k(const float* __restrict__ src, bf16* __restrict__ dst, int n) {
  int i = blockIdx.x * 256 + threadIdx.x;
  if (i < n) dst[i] = (bf16)src[i];
}

// C[M,N] = A[M,K] @ W[N,K]^T  (bf16 in, f32 accumulate), async-pipelined LDS double buffer.
// mode 0: scatter bf16 into q/k/v [B*H, N, 64] buffers (QKV projection, N=2304)
// mode 1: f32 out + bias (proj, N=768)
__global__ __launch_bounds__(256) void gemm_bf16_k(
    const bf16* __restrict__ A, const bf16* __restrict__ W,
    int M, int Nn, int K, int mode,
    bf16* __restrict__ qo, bf16* __restrict__ ko, bf16* __restrict__ vo,
    const float* __restrict__ bias, float* __restrict__ fout)
{
  __shared__ bf16 As[2][128][40];   // 80B rows: 16B-aligned for async B128
  __shared__ bf16 Ws[2][128][40];
  const int tid = threadIdx.x, lane = tid & 31, wid = tid >> 5;
  const int wm = wid & 3, wn = wid >> 2;            // wave tile: 32 rows x 64 cols
  const int m0 = blockIdx.y * 128, n0 = blockIdx.x * 128;

  v8f vz = {};
  v8f acc[2][4];
#pragma unroll
  for (int i = 0; i < 2; ++i)
#pragma unroll
    for (int j = 0; j < 4; ++j) acc[i][j] = vz;

  // 4 async B128 copies per thread per k-tile (128x32 bf16 per operand)
  auto issue_tile = [&](int buf, int k0) {
#pragma unroll
    for (int i = 0; i < 2; ++i) {
      int c = tid + 256 * i;                 // 0..511 16B chunks
      int row = c >> 2, qd = (c & 3) * 8;    // 128 rows x 4 chunks
      async_b128(&As[buf][row][qd], A, (unsigned)(((m0 + row) * K + k0 + qd) * 2));
      async_b128(&Ws[buf][row][qd], W, (unsigned)(((n0 + row) * K + k0 + qd) * 2));
    }
  };

  const int nk = K >> 5;
  issue_tile(0, 0);
  for (int kt = 0; kt < nk; ++kt) {
    if (kt + 1 < nk) {
      issue_tile((kt + 1) & 1, (kt + 1) * 32);  // prefetch next tile into alt buffer
      WAIT_ASYNC("0x4");                        // current tile's 4 loads complete
    } else {
      WAIT_ASYNC("0x0");
    }
    __syncthreads();
    const int pb = kt & 1;
    v16bf af[2], bfr[4];
#pragma unroll
    for (int mt = 0; mt < 2; ++mt) af[mt] = a_frag(&As[pb][wm * 32 + mt * 16][0], 40, 0, lane);
#pragma unroll
    for (int nt = 0; nt < 4; ++nt) bfr[nt] = b_frag_nk(&Ws[pb][wn * 64 + nt * 16][0], 40, 0, lane);
#pragma unroll
    for (int mt = 0; mt < 2; ++mt)
#pragma unroll
      for (int nt = 0; nt < 4; ++nt)
        acc[mt][nt] = WMMA_BF16(af[mt], bfr[nt], acc[mt][nt]);
    __syncthreads();
  }

#pragma unroll
  for (int mt = 0; mt < 2; ++mt)
#pragma unroll
    for (int nt = 0; nt < 4; ++nt)
#pragma unroll
      for (int r = 0; r < 8; ++r) {
        int mg = m0 + wm * 32 + mt * 16 + (lane >> 4) * 8 + r;
        int ng = n0 + wn * 64 + nt * 16 + (lane & 15);
        float val = acc[mt][nt][r];
        if (mode == 0) {
          int which = ng / 768, rem = ng % 768;
          int h = rem >> 6, d = rem & 63;
          int b = mg >> 10, nn = mg & 1023;
          bf16* dst = which == 0 ? qo : (which == 1 ? ko : vo);
          dst[(size_t)((b * 12 + h) * 1024 + nn) * 64 + d] = (bf16)val;
        } else {
          fout[(size_t)mg * Nn + ng] = val + bias[ng];
        }
      }
}

// Fused self-correlation attention, flash-style online softmax.
// grid = (B*H=48, N/128=8); 8 waves, each wave owns 16 query rows.
// Two passes over keys: pass0 maps {vv, kk}, pass1 maps {qq, qk(ori)}.
// Key-block staging is async double-buffered (ASYNCcnt pipelined).
__global__ __launch_bounds__(256) void flash_selfcor_k(
    const bf16* __restrict__ q, const bf16* __restrict__ k, const bf16* __restrict__ v,
    bf16* __restrict__ ys, bf16* __restrict__ yo)
{
  __shared__ bf16 cQ[2][64][72], cK[2][64][72], cV[2][64][72];  // 144B rows: 16B-aligned
  __shared__ bf16 Pst[8][16 * 72];
  const int tid = threadIdx.x, lane = tid & 31, wid = tid >> 5;
  const int bh = blockIdx.x, mb = blockIdx.y;
  const int b = bh / 12, h = bh % 12;
  const size_t hoff = (size_t)bh * 1024 * 64;
  const bf16 *qh = q + hoff, *kh = k + hoff, *vh = v + hoff;
  const int r0 = mb * 128 + wid * 16;
  bf16* pb = &Pst[wid][0];

  // 6 async B128 copies per thread per key block (3 x 64x64 bf16)
  auto issue_cols = [&](int buf, int kb) {
#pragma unroll
    for (int i = 0; i < 2; ++i) {
      int c = tid + 256 * i;                 // 0..511 16B chunks
      int row = c >> 3, qd = (c & 7) * 8;    // 64 rows x 8 chunks
      unsigned g = (unsigned)(((kb * 64 + row) * 64 + qd) * 2);
      async_b128(&cQ[buf][row][qd], qh, g);
      async_b128(&cK[buf][row][qd], kh, g);
      async_b128(&cV[buf][row][qd], vh, g);
    }
  };

  v8f vz = {};
  v8f ysum[4];
#pragma unroll
  for (int dt = 0; dt < 4; ++dt) ysum[dt] = vz;

  for (int pass = 0; pass < 2; ++pass) {
    const bf16* rowm[2];
    int sel0, sel1;                       // col-matrix selector: 0=Q,1=K,2=V
    if (pass == 0) { rowm[0] = vh; rowm[1] = kh; sel0 = 2; sel1 = 1; }
    else           { rowm[0] = qh; rowm[1] = qh; sel0 = 0; sel1 = 1; }

    v16bf ar[2][2];
#pragma unroll
    for (int mi = 0; mi < 2; ++mi)
#pragma unroll
      for (int c = 0; c < 2; ++c)
        ar[mi][c] = a_frag(rowm[mi] + (size_t)r0 * 64, 64, c * 32, lane);

    float mrow[2][8], lrow[2][8];
    v8f O[2][4];
#pragma unroll
    for (int mi = 0; mi < 2; ++mi) {
#pragma unroll
      for (int r = 0; r < 8; ++r) { mrow[mi][r] = -1e30f; lrow[mi][r] = 0.f; }
#pragma unroll
      for (int dt = 0; dt < 4; ++dt) O[mi][dt] = vz;
    }

    issue_cols(0, 0);
    for (int kb = 0; kb < 16; ++kb) {
      if (kb + 1 < 16) {
        issue_cols((kb + 1) & 1, kb + 1);  // prefetch next key block into alt buffer
        WAIT_ASYNC("0x6");                 // current block's 6 loads complete
      } else {
        WAIT_ASYNC("0x0");
      }
      __syncthreads();
      const int pbuf = kb & 1;
      const bf16* cbase[3] = { &cQ[pbuf][0][0], &cK[pbuf][0][0], &cV[pbuf][0][0] };
      const bf16* colm[2] = { cbase[sel0], cbase[sel1] };
      const bf16* cvb = cbase[2];

#pragma unroll
      for (int mi = 0; mi < 2; ++mi) {
        // S = rowA @ colB^T  (16 x 64 logits)
        v8f s[4];
#pragma unroll
        for (int nt = 0; nt < 4; ++nt) {
          s[nt] = vz;
#pragma unroll
          for (int c = 0; c < 2; ++c) {
            v16bf bfr = b_frag_nk(colm[mi] + nt * 16 * 72, 72, c * 32, lane);
            s[nt] = WMMA_BF16(ar[mi][c], bfr, s[nt]);
          }
        }
#pragma unroll
        for (int nt = 0; nt < 4; ++nt)
#pragma unroll
          for (int r = 0; r < 8; ++r) s[nt][r] *= 0.125f;  // hd^-0.5

        // online softmax (row stats reduced across the 16-lane half)
        float tm[8], alpha[8], rs[8];
#pragma unroll
        for (int r = 0; r < 8; ++r)
          tm[r] = fmaxf(fmaxf(s[0][r], s[1][r]), fmaxf(s[2][r], s[3][r]));
#pragma unroll
        for (int st = 1; st < 16; st <<= 1)
#pragma unroll
          for (int r = 0; r < 8; ++r) tm[r] = fmaxf(tm[r], __shfl_xor(tm[r], st, 32));
#pragma unroll
        for (int r = 0; r < 8; ++r) {
          float nm = fmaxf(mrow[mi][r], tm[r]);
          alpha[r] = __expf(mrow[mi][r] - nm);
          mrow[mi][r] = nm;
          lrow[mi][r] *= alpha[r];
          rs[r] = 0.f;
        }
#pragma unroll
        for (int nt = 0; nt < 4; ++nt)
#pragma unroll
          for (int r = 0; r < 8; ++r) {
            float p = __expf(s[nt][r] - mrow[mi][r]);
            s[nt][r] = p;
            rs[r] += p;
          }
#pragma unroll
        for (int st = 1; st < 16; st <<= 1)
#pragma unroll
          for (int r = 0; r < 8; ++r) rs[r] += __shfl_xor(rs[r], st, 32);
#pragma unroll
        for (int r = 0; r < 8; ++r) lrow[mi][r] += rs[r];
#pragma unroll
        for (int dt = 0; dt < 4; ++dt)
#pragma unroll
          for (int r = 0; r < 8; ++r) O[mi][dt][r] *= alpha[r];

        // P: C-layout -> LDS -> A-layout (per-wave region; in-wave ordering)
#pragma unroll
        for (int nt = 0; nt < 4; ++nt)
#pragma unroll
          for (int r = 0; r < 8; ++r)
            pb[((lane >> 4) * 8 + r) * 72 + nt * 16 + (lane & 15)] = (bf16)s[nt][r];

        // O += P @ V_block
#pragma unroll
        for (int c = 0; c < 2; ++c) {
          v16bf pa = a_frag(pb, 72, c * 32, lane);
#pragma unroll
          for (int dt = 0; dt < 4; ++dt) {
            v16bf bv = b_frag_kn(cvb, 72, c * 32, dt * 16, lane);
            O[mi][dt] = WMMA_BF16(pa, bv, O[mi][dt]);
          }
        }
      }
      __syncthreads();
    }

    // fold this pass's maps
#pragma unroll
    for (int mi = 0; mi < 2; ++mi) {
      bool is_ori = (pass == 1 && mi == 1);
#pragma unroll
      for (int dt = 0; dt < 4; ++dt)
#pragma unroll
        for (int r = 0; r < 8; ++r) {
          float val = O[mi][dt][r] / lrow[mi][r];
          if (is_ori) {
            int row = r0 + (lane >> 4) * 8 + r;
            int d = dt * 16 + (lane & 15);
            yo[(size_t)(b * 1024 + row) * 768 + h * 64 + d] = (bf16)val;
          } else {
            ysum[dt][r] += val;
          }
        }
    }
  }

#pragma unroll
  for (int dt = 0; dt < 4; ++dt)
#pragma unroll
    for (int r = 0; r < 8; ++r) {
      int row = r0 + (lane >> 4) * 8 + r;
      int d = dt * 16 + (lane & 15);
      ys[(size_t)(b * 1024 + row) * 768 + h * 64 + d] = (bf16)ysum[dt][r];
    }
}

extern "C" void kernel_launch(void* const* d_in, const int* in_sizes, int n_in,
                              void* d_out, int out_size, void* d_ws, size_t ws_size,
                              hipStream_t stream) {
  (void)in_sizes; (void)n_in; (void)out_size; (void)ws_size;
  const float* x     = (const float*)d_in[0];   // [4,1024,768]
  const float* Wqkv  = (const float*)d_in[1];   // [2304,768]
  const float* Wproj = (const float*)d_in[2];   // [768,768]
  const float* bproj = (const float*)d_in[3];   // [768]
  float* out = (float*)d_out;                   // x_out then x_ori

  char* p = (char*)d_ws;
  auto alloc = [&](size_t bytes) { char* r = p; p += (bytes + 255) & ~(size_t)255; return r; };
  bf16* xb  = (bf16*)alloc((size_t)4096 * 768 * 2);
  bf16* wqb = (bf16*)alloc((size_t)2304 * 768 * 2);
  bf16* wpb = (bf16*)alloc((size_t)768 * 768 * 2);
  bf16* qb  = (bf16*)alloc((size_t)48 * 1024 * 64 * 2);
  bf16* kb  = (bf16*)alloc((size_t)48 * 1024 * 64 * 2);
  bf16* vb  = (bf16*)alloc((size_t)48 * 1024 * 64 * 2);
  bf16* ysb = (bf16*)alloc((size_t)4096 * 768 * 2);
  bf16* yob = (bf16*)alloc((size_t)4096 * 768 * 2);

  cvt_bf16_k<<<(3145728 + 255) / 256, 256, 0, stream>>>(x, xb, 3145728);
  cvt_bf16_k<<<(1769472 + 255) / 256, 256, 0, stream>>>(Wqkv, wqb, 1769472);
  cvt_bf16_k<<<(589824 + 255) / 256, 256, 0, stream>>>(Wproj, wpb, 589824);

  // QKV projection: [4096,768] x [2304,768]^T -> scatter q/k/v
  gemm_bf16_k<<<dim3(18, 32), 256, 0, stream>>>(xb, wqb, 4096, 2304, 768, 0,
                                                qb, kb, vb, nullptr, nullptr);
  // Fused 4-map attention
  flash_selfcor_k<<<dim3(48, 8), 256, 0, stream>>>(qb, kb, vb, ysb, yob);
  // Output projections
  gemm_bf16_k<<<dim3(6, 32), 256, 0, stream>>>(ysb, wpb, 4096, 768, 768, 1,
                                               nullptr, nullptr, nullptr, bproj, out);
  gemm_bf16_k<<<dim3(6, 32), 256, 0, stream>>>(yob, wpb, 4096, 768, 768, 1,
                                               nullptr, nullptr, nullptr, bproj,
                                               out + (size_t)4096 * 768);
}
